// VAE_69312182223800
// MI455X (gfx1250) — compile-verified, matrix operand order checked by reading, not run on hardware
//
#include <hip/hip_runtime.h>
#include <hip/hip_bf16.h>
#include <math.h>

typedef __attribute__((ext_vector_type(16))) _Float16 v16h;
typedef __attribute__((ext_vector_type(8)))  _Float16 v8h;
typedef __attribute__((ext_vector_type(8)))  float    v8f;
typedef int v4i_gcc __attribute__((vector_size(16)));

#define NEG_SLOPE 0.01f
#define BN_EPS 1e-5f

#if defined(__has_builtin)
#if __has_builtin(__builtin_amdgcn_global_load_async_to_lds_b128) && \
    __has_builtin(__builtin_amdgcn_s_wait_asynccnt)
#define HAVE_ASYNC_LDS 1
#endif
#endif

// lrelu(x) = max(x, 0.01x) for all x (0.01x > x iff x < 0)
__device__ __forceinline__ float lrelu(float x) { return fmaxf(x, NEG_SLOPE * x); }

__device__ __forceinline__ v8f wmma32(v16h a, v16h b, v8f c) {
  return __builtin_amdgcn_wmma_f32_16x16x32_f16(false, a, false, b, (short)0, c, false, false);
}

// A fragment from row-major f16 [16 x >=32] tile, leading dim ld.
// Per-lane data is two contiguous 8-element runs -> two 16B vector loads.
__device__ __forceinline__ v16h frag_a(const _Float16* base, int ld, int lane) {
  int m = lane & 15, half = lane >> 4;
  const _Float16* p = base + m * ld + half * 8;
  union { v16h v; v8h h[2]; } u;
  u.h[0] = *(const v8h*)p;        // k = half*8 .. half*8+7
  u.h[1] = *(const v8h*)(p + 16); // k = 16+half*8 .. +7
  return u.v;
}

// B fragment from fragment-linear (pre-swizzled) storage: one 32B vector load.
__device__ __forceinline__ v16h frag_b_lin(const _Float16* blk, int lane) {
  return *(const v16h*)(blk + lane * 16);
}

// float atomic max via signed-max / unsigned-min bit trick (init must be -inf)
__device__ __forceinline__ void atomicMaxF(float* addr, float val) {
  if (val >= 0.f) atomicMax((int*)addr, __float_as_int(val));
  else            atomicMin((unsigned int*)addr, (unsigned int)__float_as_int(val));
}

// ---------------- prep kernels ----------------
__global__ void k_cvt_f16(const float* __restrict__ src, _Float16* __restrict__ dst, int n) {
  int i = blockIdx.x * 256 + threadIdx.x;
  if (i < n) dst[i] = (_Float16)src[i];
}
__global__ void k_fill_neginf(float* __restrict__ p, int n) {
  int i = blockIdx.x * 256 + threadIdx.x;
  if (i < n) p[i] = -INFINITY;
}
// Swizzle W [Ksrc x N] f32 (row-major) into fragment-linear f16 with K padded rows.
// dst index: (((nt*(K/32) + ks)*32 + lane)*16 + j)
__global__ void k_swz_b(const float* __restrict__ W, _Float16* __restrict__ dst,
                        int Ksrc, int K, int N) {
  int i = blockIdx.x * 256 + threadIdx.x;
  int total = K * N;
  if (i >= total) return;
  int j    = i & 15;
  int lane = (i >> 4) & 31;
  int blk  = i >> 9;
  int kblocks = K >> 5;
  int ks = blk % kblocks;
  int nt = blk / kblocks;
  int half = lane >> 4, n = lane & 15;
  int v = j >> 1, odd = j & 1;
  int kk = (v < 4) ? (half * 8 + 2 * v) : (16 + half * 8 + 2 * (v - 4));
  int k = ks * 32 + kk + odd;
  int col = nt * 16 + n;
  dst[i] = (k < Ksrc) ? (_Float16)W[(size_t)k * N + col] : (_Float16)0.f;
}

// ---------------- fused pointnet MLP + masked max-pool ----------------
// grid: (N/512, B); block 256 (8 waves). Wave w handles 64 points (4 M-tiles).
__global__ __launch_bounds__(256)
void k_pointnet(const float* __restrict__ x, const int* __restrict__ lengths,
                const _Float16* __restrict__ W1s, const float* __restrict__ b1,
                const _Float16* __restrict__ W2s, const float* __restrict__ b2,
                const _Float16* __restrict__ W3s, const float* __restrict__ b3,
                const _Float16* __restrict__ W4s, const float* __restrict__ b4,
                float* __restrict__ pooled) {
  __shared__ __align__(32) _Float16 sW1[32 * 64];        // 4 frag blocks
  __shared__ __align__(32) _Float16 sW2[64 * 64];        // 8 frag blocks
  __shared__ __align__(32) _Float16 sW3[64 * 128];       // 16 frag blocks
  __shared__ __align__(32) _Float16 sH[8][2][16 * 128];  // per-wave ping/pong
  __shared__ __align__(32) _Float16 sW4[2][2048];        // one n-tile (4 frag blocks), x2
  __shared__ float sPool[1024];

  const int tid = threadIdx.x;
  const int wave = tid >> 5, l = tid & 31;
  const int m = l & 15, half = l >> 4;
  const int b = blockIdx.y;
  const int pbase = blockIdx.x * 512 + wave * 64;
  const int len = lengths[b];

  for (int i = tid; i < 32 * 64; i += 256) sW1[i] = W1s[i];
  for (int i = tid; i < 64 * 64; i += 256) sW2[i] = W2s[i];
  for (int i = tid; i < 64 * 128; i += 256) sW3[i] = W3s[i];
  for (int i = tid; i < 1024; i += 256) sPool[i] = -INFINITY;
  __syncthreads();

  _Float16* h0 = &sH[wave][0][0];
  _Float16* h1 = &sH[wave][1][0];

  v16h a4[4][4];   // [m-tile][k-step] A fragments for layer 4 (K=128)

  for (int t = 0; t < 4; ++t) {
    // ---- layer 1: [16x3pad32] @ [32x64] ----
    v16h a1;
    {
      const float* xr = x + ((size_t)b * 2048 + (pbase + t * 16 + m)) * 3;
#pragma unroll
      for (int v = 0; v < 8; ++v) {
        int k = (v < 4) ? (half * 8 + 2 * v) : (16 + half * 8 + 2 * (v - 4));
        a1[2 * v]     = (k < 3)     ? (_Float16)xr[k]     : (_Float16)0.f;
        a1[2 * v + 1] = (k + 1 < 3) ? (_Float16)xr[k + 1] : (_Float16)0.f;
      }
    }
#pragma unroll
    for (int nt = 0; nt < 4; ++nt) {
      v8f acc = {};
      acc = wmma32(a1, frag_b_lin(sW1 + nt * 512, l), acc);
      int n = nt * 16 + m;
      float bi = b1[n];
#pragma unroll
      for (int v = 0; v < 8; ++v)
        h0[(v + half * 8) * 64 + n] = (_Float16)lrelu(acc[v] + bi);
    }
    __syncthreads();

    // ---- layer 2: [16x64] @ [64x64] ----
    v16h a2[2];
    a2[0] = frag_a(h0, 64, l);
    a2[1] = frag_a(h0 + 32, 64, l);
#pragma unroll
    for (int nt = 0; nt < 4; ++nt) {
      v8f acc = {};
      acc = wmma32(a2[0], frag_b_lin(sW2 + (nt * 2 + 0) * 512, l), acc);
      acc = wmma32(a2[1], frag_b_lin(sW2 + (nt * 2 + 1) * 512, l), acc);
      int n = nt * 16 + m;
      float bi = b2[n];
#pragma unroll
      for (int v = 0; v < 8; ++v)
        h1[(v + half * 8) * 64 + n] = (_Float16)lrelu(acc[v] + bi);
    }
    __syncthreads();

    // ---- layer 3: [16x64] @ [64x128] ----
    v16h a3[2];
    a3[0] = frag_a(h1, 64, l);
    a3[1] = frag_a(h1 + 32, 64, l);
#pragma unroll
    for (int nt = 0; nt < 8; ++nt) {
      v8f acc = {};
      acc = wmma32(a3[0], frag_b_lin(sW3 + (nt * 2 + 0) * 512, l), acc);
      acc = wmma32(a3[1], frag_b_lin(sW3 + (nt * 2 + 1) * 512, l), acc);
      int n = nt * 16 + m;
      float bi = b3[n];
#pragma unroll
      for (int v = 0; v < 8; ++v)
        h0[(v + half * 8) * 128 + n] = (_Float16)lrelu(acc[v] + bi);
    }
    __syncthreads();

#pragma unroll
    for (int ks = 0; ks < 4; ++ks) a4[t][ks] = frag_a(h0 + ks * 32, 128, l);
    __syncthreads();
  }

  // ---- layer 4: [64x128] @ [128x1024], W4 streamed through LDS ----
  auto stage = [&](int nt, int buf) {
#if defined(HAVE_ASYNC_LDS)
    __builtin_amdgcn_global_load_async_to_lds_b128(
        (__attribute__((address_space(1))) v4i_gcc*)((_Float16*)W4s + (size_t)nt * 2048 + tid * 8),
        (__attribute__((address_space(3))) v4i_gcc*)&sW4[buf][tid * 8], 0, 0);
#else
    *(uint4*)&sW4[buf][tid * 8] = *(const uint4*)(W4s + (size_t)nt * 2048 + tid * 8);
#endif
  };
  stage(0, 0);

  for (int nt = 0; nt < 64; ++nt) {
#if defined(HAVE_ASYNC_LDS)
    __builtin_amdgcn_s_wait_asynccnt(0);   // staged async writes to LDS done
#endif
    __syncthreads();                        // staged tile nt visible to all
    if (nt + 1 < 64) stage(nt + 1, (nt + 1) & 1);
    const _Float16* wt = sW4[nt & 1];
    v8f acc[4] = {{}, {}, {}, {}};
#pragma unroll
    for (int ks = 0; ks < 4; ++ks) {
      v16h bf = frag_b_lin(wt + ks * 512, l);
#pragma unroll
      for (int mt = 0; mt < 4; ++mt) acc[mt] = wmma32(a4[mt][ks], bf, acc[mt]);
    }
    // Masked max over raw accumulators first; bias is per-column (uniform per
    // lane) and lrelu is monotonic, so bias+lrelu can be applied after the max.
    float mx = -INFINITY;
#pragma unroll
    for (int mt = 0; mt < 4; ++mt)
#pragma unroll
      for (int v = 0; v < 8; ++v) {
        int row = mt * 16 + v + half * 8;
        mx = fmaxf(mx, (pbase + row < len) ? acc[mt][v] : -INFINITY);
      }
    mx = fmaxf(mx, __shfl_xor(mx, 16, 32));
    if (half == 0) {
      float val = lrelu(mx + b4[nt * 16 + m]);
      atomicMaxF(&sPool[nt * 16 + m], val);
    }
  }
  __syncthreads();
  for (int c = tid; c < 1024; c += 256)
    atomicMaxF(&pooled[(size_t)b * 1024 + c], sPool[c]);
}

// ---------------- encoder head: e = lrelu(pooled @ Wf + bf) ----------------
__global__ __launch_bounds__(32)
void k_enc_head(const _Float16* __restrict__ pooled16, const _Float16* __restrict__ Wfs,
                const float* __restrict__ bf, _Float16* __restrict__ e16) {
  int l = threadIdx.x;
  int nt = blockIdx.x, mt = blockIdx.y;
  const _Float16* A = pooled16 + (size_t)mt * 16 * 1024;
  v8f acc = {};
  for (int ks = 0; ks < 32; ++ks)
    acc = wmma32(frag_a(A + ks * 32, 1024, l),
                 frag_b_lin(Wfs + ((size_t)nt * 32 + ks) * 512, l), acc);
  int n = nt * 16 + (l & 15), half = l >> 4;
  float bi = bf[n];
#pragma unroll
  for (int v = 0; v < 8; ++v) {
    int row = mt * 16 + v + half * 8;
    e16[(size_t)row * 1024 + n] = (_Float16)lrelu(acc[v] + bi);
  }
}

// ---------------- mu/logvar + reparameterize ----------------
__global__ __launch_bounds__(32)
void k_mu_lv(const _Float16* __restrict__ e16,
             const _Float16* __restrict__ Wmus, const float* __restrict__ bmu,
             const _Float16* __restrict__ Wlvs, const float* __restrict__ blv,
             const float* __restrict__ eps,
             float* __restrict__ dout, _Float16* __restrict__ z16) {
  int l = threadIdx.x;
  int nt = blockIdx.x, mt = blockIdx.y;
  const _Float16* A = e16 + (size_t)mt * 16 * 1024;
  v8f am = {}, av = {};
  for (int ks = 0; ks < 32; ++ks) {
    v16h a = frag_a(A + ks * 32, 1024, l);
    am = wmma32(a, frag_b_lin(Wmus + ((size_t)nt * 32 + ks) * 512, l), am);
    av = wmma32(a, frag_b_lin(Wlvs + ((size_t)nt * 32 + ks) * 512, l), av);
  }
  int n = nt * 16 + (l & 15), half = l >> 4;
  float bm = bmu[n], bl = blv[n];
  float* muOut = dout + 128 * 128 * 3;
  float* lvOut = dout + 128 * 128 * 3 + 128 * 256;
#pragma unroll
  for (int v = 0; v < 8; ++v) {
    int row = mt * 16 + v + half * 8;
    float mu = am[v] + bm;
    float lv = av[v] + bl;
    muOut[(size_t)row * 256 + n] = mu;
    lvOut[(size_t)row * 256 + n] = lv;
    float z = mu + eps[(size_t)row * 256 + n] * expf(0.5f * lv);
    z16[(size_t)row * 256 + n] = (_Float16)z;
  }
}

// ---------------- decoder Linear + BatchNorm + LeakyReLU ----------------
__global__ __launch_bounds__(256)
void k_dec_bn(const _Float16* __restrict__ X, int K,
              const _Float16* __restrict__ Ws, int N,
              const float* __restrict__ bias, const float* __restrict__ gamma,
              const float* __restrict__ beta, _Float16* __restrict__ out) {
  __shared__ float sSum[16], sSq[16];
  int tid = threadIdx.x, wave = tid >> 5, l = tid & 31;
  int nt = blockIdx.x;
  if (tid < 16) { sSum[tid] = 0.f; sSq[tid] = 0.f; }
  const _Float16* A = X + (size_t)wave * 16 * K;
  int kblocks = K >> 5;
  v8f acc = {};
  for (int ks = 0; ks < kblocks; ++ks)
    acc = wmma32(frag_a(A + ks * 32, K, l),
                 frag_b_lin(Ws + ((size_t)nt * kblocks + ks) * 512, l), acc);
  int n = nt * 16 + (l & 15), half = l >> 4;
  float bi = bias[n];
  float h[8], s = 0.f, sq = 0.f;
#pragma unroll
  for (int v = 0; v < 8; ++v) {
    h[v] = acc[v] + bi;
    s += h[v];
    sq += h[v] * h[v];
  }
  s  += __shfl_xor(s, 16, 32);
  sq += __shfl_xor(sq, 16, 32);
  __syncthreads();
  if (half == 0) { atomicAdd(&sSum[l & 15], s); atomicAdd(&sSq[l & 15], sq); }
  __syncthreads();
  float mean = sSum[l & 15] * (1.f / 128.f);
  float var  = sSq[l & 15] * (1.f / 128.f) - mean * mean;
  float rstd = rsqrtf(var + BN_EPS);
  float g = gamma[n], be = beta[n];
#pragma unroll
  for (int v = 0; v < 8; ++v) {
    int row = wave * 16 + v + half * 8;
    out[(size_t)row * N + n] = (_Float16)lrelu((h[v] - mean) * rstd * g + be);
  }
}

// ---------------- final linear: y = d2 @ Wd3 + bd3 ----------------
__global__ __launch_bounds__(32)
void k_dec_out(const _Float16* __restrict__ d2, const _Float16* __restrict__ Wd3s,
               const float* __restrict__ bd3, float* __restrict__ y) {
  int l = threadIdx.x;
  int nt = blockIdx.x, mt = blockIdx.y;
  const _Float16* A = d2 + (size_t)mt * 16 * 1024;
  v8f acc = {};
  for (int ks = 0; ks < 32; ++ks)
    acc = wmma32(frag_a(A + ks * 32, 1024, l),
                 frag_b_lin(Wd3s + ((size_t)nt * 32 + ks) * 512, l), acc);
  int n = nt * 16 + (l & 15), half = l >> 4;
  float bi = bd3[n];
#pragma unroll
  for (int v = 0; v < 8; ++v) {
    int row = mt * 16 + v + half * 8;
    y[(size_t)row * 384 + n] = acc[v] + bi;
  }
}

extern "C" void kernel_launch(void* const* d_in, const int* in_sizes, int n_in,
                              void* d_out, int out_size, void* d_ws, size_t ws_size,
                              hipStream_t stream) {
  const float* x       = (const float*)d_in[0];
  const int*   lengths = (const int*)d_in[1];
  const float* eps     = (const float*)d_in[2];
  const float* W1  = (const float*)d_in[3];  const float* b1  = (const float*)d_in[4];
  const float* W2  = (const float*)d_in[5];  const float* b2  = (const float*)d_in[6];
  const float* W3  = (const float*)d_in[7];  const float* b3  = (const float*)d_in[8];
  const float* W4  = (const float*)d_in[9];  const float* b4  = (const float*)d_in[10];
  const float* Wf  = (const float*)d_in[11]; const float* bf  = (const float*)d_in[12];
  const float* Wmu = (const float*)d_in[13]; const float* bmu = (const float*)d_in[14];
  const float* Wlv = (const float*)d_in[15]; const float* blv = (const float*)d_in[16];
  const float* Wd1 = (const float*)d_in[17]; const float* bd1 = (const float*)d_in[18];
  const float* g1  = (const float*)d_in[19]; const float* be1 = (const float*)d_in[20];
  const float* Wd2 = (const float*)d_in[21]; const float* bd2 = (const float*)d_in[22];
  const float* g2  = (const float*)d_in[23]; const float* be2 = (const float*)d_in[24];
  const float* Wd3 = (const float*)d_in[25]; const float* bd3 = (const float*)d_in[26];

  char* w = (char*)d_ws;
  size_t off = 0;
  auto alloc = [&](size_t bytes) -> char* {
    char* p = w + off;
    off = (off + bytes + 255) & ~(size_t)255;
    return p;
  };
  _Float16* W1s      = (_Float16*)alloc(32 * 64 * 2);
  _Float16* W2s      = (_Float16*)alloc(64 * 64 * 2);
  _Float16* W3s      = (_Float16*)alloc(64 * 128 * 2);
  _Float16* W4s      = (_Float16*)alloc(128 * 1024 * 2);
  _Float16* Wfs      = (_Float16*)alloc(1024 * 1024 * 2);
  _Float16* Wmus     = (_Float16*)alloc(1024 * 256 * 2);
  _Float16* Wlvs     = (_Float16*)alloc(1024 * 256 * 2);
  _Float16* Wd1s     = (_Float16*)alloc(256 * 1024 * 2);
  _Float16* Wd2s     = (_Float16*)alloc(1024 * 1024 * 2);
  _Float16* Wd3s     = (_Float16*)alloc(1024 * 384 * 2);
  float*    pooledF  = (float*)alloc(128 * 1024 * 4);
  _Float16* pooled16 = (_Float16*)alloc(128 * 1024 * 2);
  _Float16* e16      = (_Float16*)alloc(128 * 1024 * 2);
  _Float16* z16      = (_Float16*)alloc(128 * 256 * 2);
  _Float16* d1_16    = (_Float16*)alloc(128 * 1024 * 2);
  _Float16* d2_16    = (_Float16*)alloc(128 * 1024 * 2);

  auto swz = [&](const float* s, _Float16* d, int Ksrc, int K, int N) {
    int total = K * N;
    k_swz_b<<<(total + 255) / 256, 256, 0, stream>>>(s, d, Ksrc, K, N);
  };
  swz(W1, W1s, 3, 32, 64);
  swz(W2, W2s, 64, 64, 64);
  swz(W3, W3s, 64, 64, 128);
  swz(W4, W4s, 128, 128, 1024);
  swz(Wf, Wfs, 1024, 1024, 1024);
  swz(Wmu, Wmus, 1024, 1024, 256);
  swz(Wlv, Wlvs, 1024, 1024, 256);
  swz(Wd1, Wd1s, 256, 256, 1024);
  swz(Wd2, Wd2s, 1024, 1024, 1024);
  swz(Wd3, Wd3s, 1024, 1024, 384);
  k_fill_neginf<<<(128 * 1024 + 255) / 256, 256, 0, stream>>>(pooledF, 128 * 1024);

  k_pointnet<<<dim3(2048 / 512, 128), 256, 0, stream>>>(
      x, lengths, W1s, b1, W2s, b2, W3s, b3, W4s, b4, pooledF);

  k_cvt_f16<<<(128 * 1024 + 255) / 256, 256, 0, stream>>>(pooledF, pooled16, 128 * 1024);

  k_enc_head<<<dim3(64, 8), 32, 0, stream>>>(pooled16, Wfs, bf, e16);

  k_mu_lv<<<dim3(16, 8), 32, 0, stream>>>(e16, Wmus, bmu, Wlvs, blv, eps,
                                          (float*)d_out, z16);

  k_dec_bn<<<64, 256, 0, stream>>>(z16, 256, Wd1s, 1024, bd1, g1, be1, d1_16);
  k_dec_bn<<<64, 256, 0, stream>>>(d1_16, 1024, Wd2s, 1024, bd2, g2, be2, d2_16);

  k_dec_out<<<dim3(24, 8), 32, 0, stream>>>(d2_16, Wd3s, bd3, (float*)d_out);
}
